// EGT_Layer_50259707297792
// MI455X (gfx1250) — matrix operand order repaired and checked by reading, not compiled, not run on hardware
//
#include <hip/hip_runtime.h>

typedef __attribute__((ext_vector_type(16))) _Float16 v16h;
typedef __attribute__((ext_vector_type(8)))  float    v8f;

// Fragment = 8 VGPRs; per documented CDNA5 layouts each lane's data is two
// contiguous 16-byte runs when the source is K-contiguous.
union FragH { v16h v; uint4 q[2]; };

static __device__ inline v8f wmma_f16(v16h a, v16h b, v8f c) {
  // v_wmma_f32_16x16x32_f16 (neg_a, A, neg_b, B, c_mod, C, reuse_a, reuse_b)
  return __builtin_amdgcn_wmma_f32_16x16x32_f16(false, a, false, b, (short)0, c, false, false);
}

// Async global->LDS 16B copy (CDNA5 GLOBAL_LOAD_ASYNC_TO_LDS_B128, ASYNCcnt).
static __device__ inline void async_copy_b128(void* lds, const void* gptr) {
  const unsigned ldsAddr = (unsigned)(size_t)lds;  // generic addr low 32 = LDS offset
  asm volatile("global_load_async_to_lds_b128 %0, %1, off"
               :: "v"(ldsAddr), "v"(gptr)
               : "memory");
}
static __device__ inline void wait_asynccnt0() {
  asm volatile("s_wait_asynccnt 0x0" ::: "memory");
}

// ---------------------------------------------------------------------------
__global__ __launch_bounds__(256) void cvt_f32_f16(const float* __restrict__ s,
                                                   _Float16* __restrict__ d, int n) {
  int i = blockIdx.x * 256 + threadIdx.x;
  if (i < n) d[i] = (_Float16)s[i];
}

// ---------------------------------------------------------------------------
// LayerNorm over C=512 -> f16
__global__ __launch_bounds__(256) void ln512_f16(const float* __restrict__ x,
                                                 const float* __restrict__ g,
                                                 const float* __restrict__ bta,
                                                 _Float16* __restrict__ y) {
  __shared__ float rs[256], rq[256];
  const int row = blockIdx.x, tid = threadIdx.x;
  const size_t base = (size_t)row * 512;
  const float x0 = x[base + tid];
  const float x1 = x[base + tid + 256];
  rs[tid] = x0 + x1;
  rq[tid] = x0 * x0 + x1 * x1;
  __syncthreads();
  for (int off = 128; off; off >>= 1) {
    if (tid < off) { rs[tid] += rs[tid + off]; rq[tid] += rq[tid + off]; }
    __syncthreads();
  }
  const float mean = rs[0] * (1.f / 512.f);
  const float var  = rq[0] * (1.f / 512.f) - mean * mean;
  const float rstd = rsqrtf(var + 1e-5f);
  y[base + tid]       = (_Float16)((x0 - mean) * rstd * g[tid] + bta[tid]);
  y[base + tid + 256] = (_Float16)((x1 - mean) * rstd * g[tid + 256] + bta[tid + 256]);
}

// ---------------------------------------------------------------------------
// Fused e-LN (EW=64) + E = e_ln@W_E + b_E ; gates = sigmoid(e_ln@W_G + b_G)
__global__ __launch_bounds__(256) void eln_eg(const float* __restrict__ e,
                                              const float* __restrict__ lng,
                                              const float* __restrict__ lnb,
                                              const float* __restrict__ WE,
                                              const float* __restrict__ bE,
                                              const float* __restrict__ WG,
                                              const float* __restrict__ bG,
                                              float* __restrict__ Eo,
                                              float* __restrict__ Go) {
  __shared__ float wEs[512], wGs[512], elnS[8][64], lw[64], lb[64], bEs[8], bGs[8];
  const int tid = threadIdx.x, lane = tid & 31, w = tid >> 5;
  for (int i = tid; i < 512; i += 256) { wEs[i] = WE[i]; wGs[i] = WG[i]; }
  if (tid < 64)       { lw[tid] = lng[tid]; lb[tid] = lnb[tid]; }
  else if (tid < 72)  bEs[tid - 64] = bE[tid - 64];
  else if (tid < 80)  bGs[tid - 72] = bG[tid - 72];
  __syncthreads();
  const size_t row = (size_t)blockIdx.x * 8 + w;
  const float x0 = e[row * 64 + 2 * lane];
  const float x1 = e[row * 64 + 2 * lane + 1];
  float s = x0 + x1, q = x0 * x0 + x1 * x1;
  for (int off = 16; off; off >>= 1) { s += __shfl_xor(s, off); q += __shfl_xor(q, off); }
  const float mean = s * (1.f / 64.f);
  const float rstd = rsqrtf(q * (1.f / 64.f) - mean * mean + 1e-5f);
  elnS[w][2 * lane]     = (x0 - mean) * rstd * lw[2 * lane] + lb[2 * lane];
  elnS[w][2 * lane + 1] = (x1 - mean) * rstd * lw[2 * lane + 1] + lb[2 * lane + 1];
  __syncthreads();
  if (lane < 16) {
    const int hh = lane & 7;
    const bool isG = lane >= 8;
    const float* Wm = isG ? wGs : wEs;
    float a = isG ? bGs[hh] : bEs[hh];
#pragma unroll 8
    for (int k = 0; k < 64; ++k) a += elnS[w][k] * Wm[k * 8 + hh];
    if (isG) Go[row * 8 + hh] = 1.f / (1.f + __expf(-a));
    else     Eo[row * 8 + hh] = a;
  }
}

// ---------------------------------------------------------------------------
// WMMA GEMM, block tile 64x64, K-step 32. A tile staged via async global->LDS
// DMA (ASYNCcnt); B tile register-transposed into LDS so all fragments are
// 2 x b128 LDS loads per lane.
// QKV_MODE scatters output into Q/K: [p][b][h][l][d], V transposed: [b][h][d][l].
template <int ACT_ELU, int HAS_RES, int OUT_F32, int OUT_F16, int QKV_MODE>
__global__ __launch_bounds__(256) void gemm_wmma_f16(
    const _Float16* __restrict__ A, const _Float16* __restrict__ B,
    const float* __restrict__ bias, const float* __restrict__ resid,
    float* __restrict__ outF, _Float16* __restrict__ outH,
    _Float16* __restrict__ qkOut, _Float16* __restrict__ vtOut,
    int M, int N, int K) {
  constexpr int BM = 64, BN = 64, BK = 32, LDA = 40, LDBT = 40;
  __shared__ _Float16 As[BM * LDA];    // [row][k]
  __shared__ _Float16 BsT[BN * LDBT];  // [col][k]  (transposed)
  const int tid = threadIdx.x, lane = tid & 31, w = tid >> 5;
  const int bm = blockIdx.y, bn = blockIdx.x;
  const int rt = w & 3, ch = w >> 2;  // wave row-tile 0..3, column half 0..1
  v8f acc0 = {}, acc1 = {};
  const int arow = tid >> 2, acol = (tid & 3) * 8;  // 64x32 A tile
  const int brow = tid >> 3, bcol = (tid & 7) * 8;  // 32x64 B tile
  const size_t aBase = (size_t)(bm * BM + arow) * K;
  const int fr = rt * 16 + (lane & 15), fh = lane >> 4, fc = lane & 15;
  const int nk = K / BK;
  for (int kb = 0; kb < nk; ++kb) {
    const int k0 = kb * BK;
    // A tile: async DMA straight into LDS (no VGPR bounce)
    async_copy_b128(&As[arow * LDA + acol], &A[aBase + k0 + acol]);
    // B tile: load + 16-bit transpose into LDS (overlaps with the async DMA)
    {
      _Float16 bl[8];
      *(uint4*)bl = *(const uint4*)&B[(size_t)(k0 + brow) * N + bn * BN + bcol];
#pragma unroll
      for (int j = 0; j < 8; ++j) BsT[(bcol + j) * LDBT + brow] = bl[j];
    }
    if (kb + 1 < nk) __builtin_prefetch(&A[aBase + k0 + BK + acol], 0, 1);
    wait_asynccnt0();
    __syncthreads();
    FragH af, f0, f1;
    af.q[0] = *(const uint4*)&As[fr * LDA + fh * 16];
    af.q[1] = *(const uint4*)&As[fr * LDA + fh * 16 + 8];
    const int c0 = (ch * 32 + fc) * LDBT, c1 = (ch * 32 + 16 + fc) * LDBT;
    f0.q[0] = *(const uint4*)&BsT[c0 + fh * 16];
    f0.q[1] = *(const uint4*)&BsT[c0 + fh * 16 + 8];
    f1.q[0] = *(const uint4*)&BsT[c1 + fh * 16];
    f1.q[1] = *(const uint4*)&BsT[c1 + fh * 16 + 8];
    acc0 = wmma_f16(af.v, f0.v, acc0);
    acc1 = wmma_f16(af.v, f1.v, acc1);
    __syncthreads();
  }
  // Epilogue: VGPR r holds M = r + 8*(lane>>4), N = lane&15
  const int row0 = bm * BM + rt * 16 + fh * 8;
  v8f accs[2] = {acc0, acc1};
#pragma unroll
  for (int t = 0; t < 2; ++t) {
    const int gcol = bn * BN + ch * 32 + t * 16 + fc;
#pragma unroll
    for (int r = 0; r < 8; ++r) {
      const int gr = row0 + r;
      const size_t gi = (size_t)gr * N + gcol;
      float v = accs[t][r] + bias[gcol];
      if (ACT_ELU) v = v > 0.f ? v : expm1f(v);
      if (HAS_RES) v += resid[gi];
      if (OUT_F32) outF[gi] = v;
      if (OUT_F16) outH[gi] = (_Float16)v;
      if (QKV_MODE) {
        const int bb = gr >> 9, l = gr & 511;
        const int d2 = gcol >> 3, h2 = gcol & 7, pp = d2 >> 6, d = d2 & 63;
        const _Float16 hv = (_Float16)v;
        if (pp < 2)
          qkOut[(((size_t)(pp * 4 + bb) * 8 + h2) * 512 + l) * 64 + d] = hv;
        else
          vtOut[(((size_t)bb * 8 + h2) * 64 + d) * 512 + l] = hv;
      }
    }
  }
}

// ---------------------------------------------------------------------------
// Attention: block per (b, h, 16 l-rows). QK^T (WMMA) -> clip + E -> H_hat;
// block softmax * gates, deg = log1p(sum gates); P@V (WMMA) with fused scale.
// Q/K in [p][b][h][l][d]; V transposed [b][h][d][l]: all fragments = 2 x b128.
__global__ __launch_bounds__(256) void egt_attn(const _Float16* __restrict__ QK,
                                                const _Float16* __restrict__ Vt,
                                                const float* __restrict__ E,
                                                const float* __restrict__ G,
                                                float* __restrict__ Hh,
                                                _Float16* __restrict__ Vatt) {
  constexpr int L = 512, H = 8, D = 64;
  __shared__ float S[16 * 512];
  __shared__ _Float16 P[16 * 512];
  __shared__ float red[256], red2[256];
  __shared__ float rowMax[16], rowSE[16], rowSG[16];
  const int tid = threadIdx.x, lane = tid & 31, w = tid >> 5;
  const int l0 = blockIdx.x * 16, hh = blockIdx.y, b = blockIdx.z;
  const int fr = lane & 15, fh = lane >> 4;
  const _Float16* Qp = QK + ((size_t)(b * H + hh) * L + l0) * D;
  const _Float16* Kp = QK + ((size_t)((4 + b) * H + hh) * L) * D;
  FragH qa0, qa1;
  qa0.q[0] = *(const uint4*)&Qp[fr * D + fh * 16];
  qa0.q[1] = *(const uint4*)&Qp[fr * D + fh * 16 + 8];
  qa1.q[0] = *(const uint4*)&Qp[fr * D + 32 + fh * 16];
  qa1.q[1] = *(const uint4*)&Qp[fr * D + 32 + fh * 16 + 8];
  // Phase 1: each wave does 4 m-tiles of 16
  for (int jt = 0; jt < 4; ++jt) {
    const int m0 = (w * 4 + jt) * 16;
    const _Float16* Kr = &Kp[(size_t)(m0 + fr) * D];
    FragH k0f, k1f;
    k0f.q[0] = *(const uint4*)&Kr[fh * 16];
    k0f.q[1] = *(const uint4*)&Kr[fh * 16 + 8];
    k1f.q[0] = *(const uint4*)&Kr[32 + fh * 16];
    k1f.q[1] = *(const uint4*)&Kr[32 + fh * 16 + 8];
    v8f acc = {};
    acc = wmma_f16(qa0.v, k0f.v, acc);
    acc = wmma_f16(qa1.v, k1f.v, acc);
#pragma unroll
    for (int r = 0; r < 8; ++r) {
      const int rl = fh * 8 + r, m = m0 + fr;
      float s = acc[r] * 0.125f;  // D^-0.5
      s = fminf(fmaxf(s, -5.f), 5.f);
      const size_t gi = ((size_t)(b * L + l0 + rl) * L + m) * H + hh;
      s += E[gi];
      Hh[gi] = s;
      S[rl * 512 + m] = s;
    }
  }
  __syncthreads();
  {  // row max
    const int row = tid >> 4, seg = tid & 15;
    float mx = -1e30f;
    for (int k = 0; k < 32; ++k) mx = fmaxf(mx, S[row * 512 + seg * 32 + k]);
    red[tid] = mx;
  }
  __syncthreads();
  if (tid < 16) {
    float mx = red[tid * 16];
    for (int k = 1; k < 16; ++k) mx = fmaxf(mx, red[tid * 16 + k]);
    rowMax[tid] = mx;
  }
  __syncthreads();
  {  // exp, gate, sums
    const int row = tid >> 4, seg = tid & 15;
    const float mx = rowMax[row];
    float ps = 0.f, gs = 0.f;
    for (int k = 0; k < 32; ++k) {
      const int m = seg + 16 * k;
      const float p = __expf(S[row * 512 + m] - mx);
      const float g = G[((size_t)(b * L + l0 + row) * L + m) * H + hh];
      P[row * 512 + m] = (_Float16)(p * g);
      ps += p;
      gs += g;
    }
    red[tid] = ps;
    red2[tid] = gs;
  }
  __syncthreads();
  if (tid < 16) {
    float ps = 0.f, gs = 0.f;
    for (int k = 0; k < 16; ++k) { ps += red[tid * 16 + k]; gs += red2[tid * 16 + k]; }
    rowSE[tid] = ps;
    rowSG[tid] = gs;
  }
  __syncthreads();
  // Phase 2: waves 0..3 compute 16x16 d-tiles of P @ V, K = 512
  if (w < 4) {
    const int nt = w;
    const _Float16* Vp = Vt + (((size_t)b * H + hh) * D + nt * 16 + fr) * L;
    v8f acc = {};
    for (int kc = 0; kc < 16; ++kc) {
      FragH pa, vb;
      pa.q[0] = *(const uint4*)&P[fr * 512 + kc * 32 + fh * 16];
      pa.q[1] = *(const uint4*)&P[fr * 512 + kc * 32 + fh * 16 + 8];
      vb.q[0] = *(const uint4*)&Vp[kc * 32 + fh * 16];
      vb.q[1] = *(const uint4*)&Vp[kc * 32 + fh * 16 + 8];
      acc = wmma_f16(pa.v, vb.v, acc);
    }
#pragma unroll
    for (int r = 0; r < 8; ++r) {
      const int rl = fh * 8 + r, d = nt * 16 + fr;
      const float sc = log1pf(rowSG[rl]) / rowSE[rl];  // softmax norm * deg
      Vatt[(size_t)(b * L + l0 + rl) * 512 + d * H + hh] = (_Float16)(acc[r] * sc);
    }
  }
}

// ---------------------------------------------------------------------------
// Fused: e1 = H_hat @ W_Oe + b_Oe + e (K=8, VALU) -> e1 f32 (d_out e-region);
//        e_f = LN(e1) over 64 -> f16. One wave per row.
__global__ __launch_bounds__(256) void e1_ln(const float* __restrict__ Hh,
                                             const float* __restrict__ WOe,
                                             const float* __restrict__ bOe,
                                             const float* __restrict__ e,
                                             const float* __restrict__ lng,
                                             const float* __restrict__ lnb,
                                             float* __restrict__ e1,
                                             _Float16* __restrict__ ef) {
  __shared__ float Ws[512], bs[64], lw[64], lb[64];
  const int tid = threadIdx.x, lane = tid & 31, w = tid >> 5;
  for (int i = tid; i < 512; i += 256) Ws[i] = WOe[i];
  if (tid < 64)        bs[tid] = bOe[tid];
  else if (tid < 128)  lw[tid - 64] = lng[tid - 64];
  else if (tid < 192)  lb[tid - 128] = lnb[tid - 128];
  __syncthreads();
  const size_t row = (size_t)blockIdx.x * 8 + w;
  float hv[8];
#pragma unroll
  for (int t = 0; t < 8; ++t) hv[t] = Hh[row * 8 + t];
  const int j0 = 2 * lane, j1 = 2 * lane + 1;
  float v0 = bs[j0] + e[row * 64 + j0];
  float v1 = bs[j1] + e[row * 64 + j1];
#pragma unroll
  for (int t = 0; t < 8; ++t) {
    v0 += hv[t] * Ws[t * 64 + j0];
    v1 += hv[t] * Ws[t * 64 + j1];
  }
  e1[row * 64 + j0] = v0;
  e1[row * 64 + j1] = v1;
  float s = v0 + v1, q = v0 * v0 + v1 * v1;
  for (int off = 16; off; off >>= 1) { s += __shfl_xor(s, off); q += __shfl_xor(q, off); }
  const float mean = s * (1.f / 64.f);
  const float rstd = rsqrtf(q * (1.f / 64.f) - mean * mean + 1e-5f);
  ef[row * 64 + j0] = (_Float16)((v0 - mean) * rstd * lw[j0] + lb[j0]);
  ef[row * 64 + j1] = (_Float16)((v1 - mean) * rstd * lw[j1] + lb[j1]);
}

// ---------------------------------------------------------------------------
extern "C" void kernel_launch(void* const* d_in, const int* in_sizes, int n_in,
                              void* d_out, int out_size, void* d_ws, size_t ws_size,
                              hipStream_t stream) {
  (void)in_sizes; (void)n_in; (void)out_size; (void)ws_size;
  const float* h      = (const float*)d_in[0];
  const float* e      = (const float*)d_in[1];
  const float* ln_h_g = (const float*)d_in[2];
  const float* ln_h_b = (const float*)d_in[3];
  const float* ln_e_g = (const float*)d_in[4];
  const float* ln_e_b = (const float*)d_in[5];
  const float* W_QKV  = (const float*)d_in[6];
  const float* b_QKV  = (const float*)d_in[7];
  const float* W_E    = (const float*)d_in[8];
  const float* b_E    = (const float*)d_in[9];
  const float* W_G    = (const float*)d_in[10];
  const float* b_G    = (const float*)d_in[11];
  const float* W_Oh   = (const float*)d_in[12];
  const float* b_Oh   = (const float*)d_in[13];
  const float* fln_h_g= (const float*)d_in[14];
  const float* fln_h_b= (const float*)d_in[15];
  const float* W_h1   = (const float*)d_in[16];
  const float* b_h1   = (const float*)d_in[17];
  const float* W_h2   = (const float*)d_in[18];
  const float* b_h2   = (const float*)d_in[19];
  const float* W_Oe   = (const float*)d_in[20];
  const float* b_Oe   = (const float*)d_in[21];
  const float* fln_e_g= (const float*)d_in[22];
  const float* fln_e_b= (const float*)d_in[23];
  const float* W_e1   = (const float*)d_in[24];
  const float* b_e1   = (const float*)d_in[25];
  const float* W_e2   = (const float*)d_in[26];
  const float* b_e2   = (const float*)d_in[27];

  const int B = 4, L = 512, C = 512, EW = 64, Hh = 8, NI = 1024;
  const size_t BL = (size_t)B * L;       // 2048
  const size_t BLL = (size_t)B * L * L;  // 1,048,576

  float* h_out = (float*)d_out;
  float* e_out = h_out + BL * C;  // e region of d_out (also holds e1)

  char* p = (char*)d_ws;
  auto alloc = [&](size_t bytes) {
    char* r = p;
    p += (bytes + 255) & ~(size_t)255;
    return r;
  };
  _Float16* wqkvh = (_Float16*)alloc((size_t)C * 3 * C * 2);
  _Float16* wohh  = (_Float16*)alloc((size_t)C * C * 2);
  _Float16* wh1h  = (_Float16*)alloc((size_t)C * NI * 2);
  _Float16* wh2h  = (_Float16*)alloc((size_t)NI * C * 2);
  _Float16* we1h  = (_Float16*)alloc((size_t)EW * EW * 2);
  _Float16* we2h  = (_Float16*)alloc((size_t)EW * EW * 2);
  _Float16* hln   = (_Float16*)alloc(BL * C * 2);
  _Float16* qkh   = (_Float16*)alloc((size_t)2 * B * Hh * L * 64 * 2);  // Q,K head-major
  _Float16* vth   = (_Float16*)alloc((size_t)B * Hh * 64 * L * 2);      // V transposed
  float*    Ebuf  = (float*)   alloc(BLL * Hh * 4);
  float*    Gbuf  = (float*)   alloc(BLL * Hh * 4);
  float*    Hhat  = (float*)   alloc(BLL * Hh * 4);
  _Float16* vatth = (_Float16*)alloc(BL * C * 2);
  float*    h1f   = (float*)   alloc(BL * C * 4);
  _Float16* hfh   = (_Float16*)alloc(BL * C * 2);
  _Float16* midh  = (_Float16*)alloc(BL * NI * 2);
  _Float16* efh   = (_Float16*)alloc(BLL * EW * 2);  // e_f, then e-FFN mid in place

  auto cvt = [&](const float* s, _Float16* d, int n) {
    cvt_f32_f16<<<(n + 255) / 256, 256, 0, stream>>>(s, d, n);
  };
  cvt(W_QKV, wqkvh, C * 3 * C);
  cvt(W_Oh,  wohh,  C * C);
  cvt(W_h1,  wh1h,  C * NI);
  cvt(W_h2,  wh2h,  NI * C);
  cvt(W_e1,  we1h,  EW * EW);
  cvt(W_e2,  we2h,  EW * EW);

  ln512_f16<<<(int)BL, 256, 0, stream>>>(h, ln_h_g, ln_h_b, hln);
  eln_eg<<<(int)(BLL / 8), 256, 0, stream>>>(e, ln_e_g, ln_e_b, W_E, b_E, W_G, b_G,
                                             Ebuf, Gbuf);
  // QKV GEMM, scattered into head-major Q/K and transposed V
  gemm_wmma_f16<0, 0, 0, 0, 1><<<dim3(3 * C / 64, (int)BL / 64), 256, 0, stream>>>(
      hln, wqkvh, b_QKV, nullptr, nullptr, nullptr, qkh, vth, (int)BL, 3 * C, C);
  // attention -> H_hat, V_att (softmax norm & deg fused)
  egt_attn<<<dim3(L / 16, Hh, B), 256, 0, stream>>>(qkh, vth, Ebuf, Gbuf, Hhat, vatth);
  // h1 = V_att @ W_Oh + b_Oh + h
  gemm_wmma_f16<0, 1, 1, 0, 0><<<dim3(C / 64, (int)BL / 64), 256, 0, stream>>>(
      vatth, wohh, b_Oh, h, h1f, nullptr, nullptr, nullptr, (int)BL, C, C);
  // h FFN
  ln512_f16<<<(int)BL, 256, 0, stream>>>(h1f, fln_h_g, fln_h_b, hfh);
  gemm_wmma_f16<1, 0, 0, 1, 0><<<dim3(NI / 64, (int)BL / 64), 256, 0, stream>>>(
      hfh, wh1h, b_h1, nullptr, nullptr, midh, nullptr, nullptr, (int)BL, NI, C);
  gemm_wmma_f16<0, 1, 1, 0, 0><<<dim3(C / 64, (int)BL / 64), 256, 0, stream>>>(
      midh, wh2h, b_h2, h1f, h_out, nullptr, nullptr, nullptr, (int)BL, C, NI);
  // e path: e1 (into d_out) + LN -> e_f
  e1_ln<<<(int)(BLL / 8), 256, 0, stream>>>(Hhat, W_Oe, b_Oe, e, fln_e_g, fln_e_b,
                                            e_out, efh);
  // e FFN (GEMM1 in-place on efh; GEMM2 accumulates residual e1 into d_out)
  gemm_wmma_f16<1, 0, 0, 1, 0><<<dim3(EW / 64, (int)(BLL / 64)), 256, 0, stream>>>(
      efh, we1h, b_e1, nullptr, nullptr, efh, nullptr, nullptr, (int)BLL, EW, EW);
  gemm_wmma_f16<0, 1, 1, 0, 0><<<dim3(EW / 64, (int)(BLL / 64)), 256, 0, stream>>>(
      efh, we2h, b_e2, e_out, e_out, nullptr, nullptr, nullptr, (int)BLL, EW, EW);
}
